// ColorJitter_18176301597035
// MI455X (gfx1250) — compile-verified, matrix operand order checked by reading, not run on hardware
//
#include <hip/hip_runtime.h>

typedef __attribute__((ext_vector_type(16))) _Float16 v16h;
typedef __attribute__((ext_vector_type(8)))  float    v8f;
typedef __attribute__((ext_vector_type(4)))  float    v4f;

#define TPB   256
#define FPT   12            // floats per thread = 4 RGB pixels
#define CHUNK (TPB * FPT)   // 3072 floats per block (multiple of 3 and 16B)

// ---------------------------------------------------------------------------
// Wave-level (32 lane) f32 sum via WMMA ones-matrix trick.
// Each lane's value goes into one f16 slot of the A matrix; B = all-ones.
// D[m,n] = rowsum(A)[m]  (independent of column), f32 accumulate.
// Sum of the 8 D VGPRs in-lane covers M=0..7 (lanes 0-15) / M=8..15
// (lanes 16-31); one xor-16 shuffle-add yields the total in every lane.
// Per-thread partials are <= 4.0 so the f16 conversion is ~2^-11 relative.
// ---------------------------------------------------------------------------
__device__ __forceinline__ float wave_sum_wmma(float x) {
  v16h a = {};
  a[0] = (_Float16)x;
  v16h ones;
#pragma unroll
  for (int i = 0; i < 16; ++i) ones[i] = (_Float16)1.0f;
  v8f c = {};
  v8f d = __builtin_amdgcn_wmma_f32_16x16x32_f16(
      /*neg_a=*/false, a, /*neg_b=*/false, ones,
      /*c_mod=*/(short)0, c, /*reuse_a=*/false, /*reuse_b=*/false);
  float s = d[0] + d[1] + d[2] + d[3] + d[4] + d[5] + d[6] + d[7];
  s += __shfl_xor(s, 16, 32);
  return s;
}

// ---------------------------------------------------------------------------
// Pass 1: per-(batch,channel) partial sums. Regular-temporal loads pull the
// image into the 192MB L2 so pass 3 re-reads hit L2 instead of HBM.
// ---------------------------------------------------------------------------
__global__ void cj_reduce(const float* __restrict__ img, float* __restrict__ part,
                          int perBatch, int bpb) {
  int b   = blockIdx.x / bpb;
  int blk = blockIdx.x % bpb;
  long base = (long)b * perBatch + (long)blk * CHUNK;
  int  off  = threadIdx.x * FPT;
  long rem  = (long)perBatch - (long)blk * CHUNK;

  float r = 0.f, g = 0.f, bl = 0.f;
  if (off + FPT <= rem) {
    const v4f* src = (const v4f*)(img + base + off);
    v4f p0 = src[0], p1 = src[1], p2 = src[2];
    r  = p0.x + p0.w + p1.z + p2.y;
    g  = p0.y + p1.x + p1.w + p2.z;
    bl = p0.z + p1.y + p2.x + p2.w;
  } else {
    int end = (int)((rem < (long)(off + FPT)) ? rem : (long)(off + FPT));
    for (int i = off; i < end; ++i) {   // CHUNK,off multiples of 3 -> i%3 = channel
      float v = img[base + i];
      int c = i % 3;
      r  += (c == 0) ? v : 0.f;
      g  += (c == 1) ? v : 0.f;
      bl += (c == 2) ? v : 0.f;
    }
  }

  float wr = wave_sum_wmma(r);
  float wg = wave_sum_wmma(g);
  float wb = wave_sum_wmma(bl);

  __shared__ float lds[(TPB / 32) * 3];
  int lane = threadIdx.x & 31, wv = threadIdx.x >> 5;
  if (lane == 0) { lds[wv*3+0] = wr; lds[wv*3+1] = wg; lds[wv*3+2] = wb; }
  __syncthreads();
  if (threadIdx.x == 0) {
    float R = 0.f, G = 0.f, Bv = 0.f;
#pragma unroll
    for (int w = 0; w < TPB / 32; ++w) {
      R += lds[w*3+0]; G += lds[w*3+1]; Bv += lds[w*3+2];
    }
    part[blockIdx.x*3+0] = R;
    part[blockIdx.x*3+1] = G;
    part[blockIdx.x*3+2] = Bv;
  }
}

// ---------------------------------------------------------------------------
// Pass 2: fold block partials per (batch,channel) in fixed order (deterministic).
// ---------------------------------------------------------------------------
__global__ void cj_finalize(const float* __restrict__ part, float* __restrict__ sums,
                            int bpb, int n) {
  int idx = blockIdx.x * blockDim.x + threadIdx.x;
  if (idx >= n) return;
  int b = idx / 3, c = idx % 3;
  float s = 0.f;
  for (int k = 0; k < bpb; ++k) s += part[(b * bpb + k) * 3 + c];
  sums[idx] = s;
}

// ---------------------------------------------------------------------------
// Color math (matches the JAX reference semantics exactly, incl. where-selects)
// ---------------------------------------------------------------------------
__device__ __forceinline__ void rgb2hsv(float r, float g, float b,
                                        float& h, float& s, float& v) {
  float mx  = fmaxf(r, fmaxf(g, b));
  float mn  = fminf(r, fminf(g, b));
  float rng = mx - mn;
  v = mx;
  s = (mx > 0.f) ? rng / mx : 0.f;
  float norm = (rng != 0.f) ? 1.f / (6.f * rng) : 1.0e9f;
  float hr = norm * (g - b);
  float hg = norm * (b - r) + (1.f / 3.f);
  float hb = norm * (r - g) + (2.f / 3.f);
  float hh = (r == mx) ? hr : ((g == mx) ? hg : hb);
  hh = (rng > 0.f) ? hh : 0.f;
  h  = (hh < 0.f) ? hh + 1.f : hh;
}

__device__ __forceinline__ void hsv2rgb(float h, float s, float v,
                                        float& r, float& g, float& b) {
  float c  = s * v;
  float m  = v - c;
  float hf = h - floorf(h);                 // h mod 1  (python % semantics)
  float dh = hf * 6.f;
  float fm = dh - 2.f * floorf(dh * 0.5f);  // dh mod 2
  float x  = c * (1.f - fabsf(fm - 1.f));
  int  hc  = (int)dh;                       // dh >= 0 -> trunc == floor
  float rr = (hc == 0 || hc == 5) ? c : ((hc == 1 || hc == 4) ? x : 0.f);
  float gg = (hc == 1 || hc == 2) ? c : ((hc == 0 || hc == 3) ? x : 0.f);
  float bb = (hc == 3 || hc == 4) ? c : ((hc == 2 || hc == 5) ? x : 0.f);
  r = rr + m; g = gg + m; b = bb + m;
}

__device__ __forceinline__ void jitter_px(float& r, float& g, float& b,
                                          float bd, float cf, float sf, float hd,
                                          float mr, float mg, float mb) {
  r += bd; g += bd; b += bd;
  r = cf * (r - mr) + mr;
  g = cf * (g - mg) + mg;
  b = cf * (b - mb) + mb;
  float h, s, v;
  rgb2hsv(r, g, b, h, s, v);
  s = fminf(fmaxf(s * sf, 0.f), 1.f);
  hsv2rgb(h, s, v, r, g, b);
  rgb2hsv(r, g, b, h, s, v);
  h = h + hd;
  h = h - floorf(h);                        // (h + hd) mod 1
  hsv2rgb(h, s, v, r, g, b);
  r = fminf(fmaxf(r, 0.f), 1.f);
  g = fminf(fmaxf(g, 0.f), 1.f);
  b = fminf(fmaxf(b, 0.f), 1.f);
}

// ---------------------------------------------------------------------------
// Pass 3: elementwise jitter. Input reads are last-use (NT) — they hit L2 from
// pass 1; output uses NT stores so 100MB of output doesn't evict the input.
// ---------------------------------------------------------------------------
__global__ void cj_apply(const float* __restrict__ img,
                         const float* __restrict__ bdp, const float* __restrict__ cfp,
                         const float* __restrict__ sfp, const float* __restrict__ hdp,
                         const float* __restrict__ sums, float* __restrict__ out,
                         int perBatch, int bpb, float invHW) {
  int b   = blockIdx.x / bpb;
  int blk = blockIdx.x % bpb;
  long base = (long)b * perBatch + (long)blk * CHUNK;
  int  off  = threadIdx.x * FPT;
  long rem  = (long)perBatch - (long)blk * CHUNK;

  float bd = bdp[b], cf = cfp[b], sf = sfp[b], hd = hdp[b];
  // mean of (image + bd) = mean(image) + bd
  float mr = sums[b*3+0] * invHW + bd;
  float mg = sums[b*3+1] * invHW + bd;
  float mb = sums[b*3+2] * invHW + bd;

  if (off + FPT <= rem) {
    const v4f* src = (const v4f*)(img + base + off);
    v4f p0 = __builtin_nontemporal_load(src);
    v4f p1 = __builtin_nontemporal_load(src + 1);
    v4f p2 = __builtin_nontemporal_load(src + 2);

    float r0 = p0.x, g0 = p0.y, b0 = p0.z;
    float r1 = p0.w, g1 = p1.x, b1 = p1.y;
    float r2 = p1.z, g2 = p1.w, b2 = p2.x;
    float r3 = p2.y, g3 = p2.z, b3 = p2.w;

    jitter_px(r0, g0, b0, bd, cf, sf, hd, mr, mg, mb);
    jitter_px(r1, g1, b1, bd, cf, sf, hd, mr, mg, mb);
    jitter_px(r2, g2, b2, bd, cf, sf, hd, mr, mg, mb);
    jitter_px(r3, g3, b3, bd, cf, sf, hd, mr, mg, mb);

    v4f q0 = {r0, g0, b0, r1};
    v4f q1 = {g1, b1, r2, g2};
    v4f q2 = {b2, r3, g3, b3};
    v4f* dst = (v4f*)(out + base + off);
    __builtin_nontemporal_store(q0, dst);
    __builtin_nontemporal_store(q1, dst + 1);
    __builtin_nontemporal_store(q2, dst + 2);
  } else if (off < rem) {
    int end = (int)((rem < (long)(off + FPT)) ? rem : (long)(off + FPT));
    for (int i = off; i + 3 <= end; i += 3) {
      float r = img[base + i], g = img[base + i + 1], bb = img[base + i + 2];
      jitter_px(r, g, bb, bd, cf, sf, hd, mr, mg, mb);
      out[base + i] = r; out[base + i + 1] = g; out[base + i + 2] = bb;
    }
  }
}

// ---------------------------------------------------------------------------
extern "C" void kernel_launch(void* const* d_in, const int* in_sizes, int n_in,
                              void* d_out, int out_size, void* d_ws, size_t ws_size,
                              hipStream_t stream) {
  const float* img = (const float*)d_in[0];
  const float* bdp = (const float*)d_in[1];
  const float* cfp = (const float*)d_in[2];
  const float* sfp = (const float*)d_in[3];
  const float* hdp = (const float*)d_in[4];
  float* out = (float*)d_out;

  int  Bn       = in_sizes[1];                    // 128 batches
  long total    = (long)in_sizes[0];              // B*H*W*3
  int  perBatch = (int)(total / Bn);              // H*W*3 = 196608
  int  hw       = perBatch / 3;                   // 65536
  int  bpb      = (perBatch + CHUNK - 1) / CHUNK; // 64 blocks per batch
  int  grid     = Bn * bpb;                       // 8192 blocks

  float* part = (float*)d_ws;                     // grid*3 partials
  float* sums = part + (size_t)grid * 3;          // Bn*3 channel sums

  cj_reduce<<<grid, TPB, 0, stream>>>(img, part, perBatch, bpb);
  cj_finalize<<<(Bn * 3 + 255) / 256, 256, 0, stream>>>(part, sums, bpb, Bn * 3);
  cj_apply<<<grid, TPB, 0, stream>>>(img, bdp, cfp, sfp, hdp, sums, out,
                                     perBatch, bpb, 1.0f / (float)hw);
}
